// Feature_fusion_45938970198292
// MI455X (gfx1250) — matrix operand order repaired and checked by reading, not compiled, not run on hardware
//
#include <hip/hip_runtime.h>
#include <hip/hip_bf16.h>
#include <math.h>

// CDNA5 / gfx1250 implementation. wave32 everywhere; all matmuls use
// v_wmma_f32_16x16x32_f16 (f16 operands, f32 accumulate).

typedef __attribute__((ext_vector_type(16))) _Float16 v16h;
typedef __attribute__((ext_vector_type(8)))  _Float16 v8h;
typedef __attribute__((ext_vector_type(8)))  float    v8f;

#define DEVINL __device__ __forceinline__

DEVINL v16h cat16(v8h a, v8h b) {
  v16h r;
#pragma unroll
  for (int i = 0; i < 8; ++i) { r[i] = a[i]; r[i + 8] = b[i]; }
  return r;
}

// A fragment (16x32, f16): rows row0..row0+15 stored row-major at `base`
// (base already offset to row0), leading dim ld (elements), K columns
// cb..cb+31.  Lane L (<16): row=L, K={cb..cb+7, cb+16..cb+23};
// lane L (>=16): row=L-16, K={cb+8..cb+15, cb+24..cb+31}.
DEVINL v16h loadA(const _Float16* base, int ld, int cb, int lr, int half) {
  const _Float16* p = base + (size_t)lr * ld + cb + half * 8;
  v8h a0 = *(const v8h*)p;
  v8h a1 = *(const v8h*)(p + 16);
  return cat16(a0, a1);
}

// B fragment (32x16, f16) from "NT" storage: B column j is stored as row
// (col0+j) of `base` (base already offset to col0).  Lane holds column
// lane%16 and K={cb+8h..cb+8h+7, cb+16+8h..cb+16+8h+7} (h = lane-half).
DEVINL v16h loadB(const _Float16* base, int ld, int cb, int lr, int half) {
  const _Float16* p = base + (size_t)lr * ld + cb + half * 8;
  v8h b0 = *(const v8h*)p;
  v8h b1 = *(const v8h*)(p + 16);
  return cat16(b0, b1);
}

DEVINL v8f wmma(v16h a, v16h b, v8f c) {
  return __builtin_amdgcn_wmma_f32_16x16x32_f16(false, a, false, b, (short)0, c,
                                                false, false);
}

// ---------------------------------------------------------------- convert
__global__ void cdna5_cvt_f16(const float* __restrict__ s,
                              _Float16* __restrict__ d, int n) {
  int i = blockIdx.x * blockDim.x + threadIdx.x;
  if (i < n) d[i] = (_Float16)s[i];
}

// ------------------------------------------------------------ proj (NT)
// OT[n,o] = sum_c X[c,n] * W[o,c];  X f32 (K x 4096), W f16 (Mout x K),
// OT f16 (4096 x Mout).  WG tile: 128n x 64o; wave w owns rows w*16..w*16+15.
__global__ __launch_bounds__(256) void cdna5_proj_nt(
    const float* __restrict__ X, const _Float16* __restrict__ W,
    _Float16* __restrict__ OT, int K, int Mout, size_t xStride, size_t oStride) {
  const int N = 4096;
  X  += (size_t)blockIdx.z * xStride;
  OT += (size_t)blockIdx.z * oStride;
  const int n0 = blockIdx.x * 128, o0 = blockIdx.y * 64;
  const int tid = threadIdx.x;
  const int wave = tid >> 5, lane = tid & 31, lr = lane & 15, half = lane >> 4;
  constexpr int LDP = 40;                 // 32 c + pad, keeps 16B alignment
  __shared__ _Float16 xs[128 * LDP];      // X chunk transposed: [n][c]

  v8f acc[4];
#pragma unroll
  for (int t = 0; t < 4; ++t)
#pragma unroll
    for (int r = 0; r < 8; ++r) acc[t][r] = 0.f;

  for (int cb0 = 0; cb0 < K; cb0 += 32) {
    __syncthreads();
#pragma unroll
    for (int i = 0; i < 16; ++i) {        // stage 32c x 128n, coalesced in n
      int idx = tid + i * 256;
      int ci = idx >> 7, nn = idx & 127;
      xs[nn * LDP + ci] = (_Float16)X[(size_t)(cb0 + ci) * N + n0 + nn];
    }
    __syncthreads();
    v16h a = loadA(xs + (wave * 16) * LDP, LDP, 0, lr, half);
#pragma unroll
    for (int ot = 0; ot < 4; ++ot) {
      v16h b = loadB(W + (size_t)(o0 + ot * 16) * K, K, cb0, lr, half);
      acc[ot] = wmma(a, b, acc[ot]);
    }
  }
  const int nb = n0 + wave * 16;
#pragma unroll
  for (int ot = 0; ot < 4; ++ot)
#pragma unroll
    for (int r = 0; r < 8; ++r)
      OT[(size_t)(nb + r + 8 * half) * Mout + o0 + ot * 16 + lr] =
          (_Float16)acc[ot][r];
}

// ------------------------------------------------------------ proj (NN)
// O[o,n] = sum_c W[o,c] * X[c,n];  O f16 (Mout x 4096)  (used for V)
__global__ __launch_bounds__(256) void cdna5_proj_nn(
    const float* __restrict__ X, const _Float16* __restrict__ W,
    _Float16* __restrict__ O, int K, size_t xStride, size_t oStride) {
  const int N = 4096;
  X += (size_t)blockIdx.z * xStride;
  O += (size_t)blockIdx.z * oStride;
  const int n0 = blockIdx.x * 128, o0 = blockIdx.y * 64;
  const int tid = threadIdx.x;
  const int wave = tid >> 5, lane = tid & 31, lr = lane & 15, half = lane >> 4;
  const int ot = wave & 3, ng = (wave >> 2) * 4;
  constexpr int LDP = 40;
  __shared__ _Float16 xs[128 * LDP];

  v8f acc[4];
#pragma unroll
  for (int t = 0; t < 4; ++t)
#pragma unroll
    for (int r = 0; r < 8; ++r) acc[t][r] = 0.f;

  for (int cb0 = 0; cb0 < K; cb0 += 32) {
    __syncthreads();
#pragma unroll
    for (int i = 0; i < 16; ++i) {
      int idx = tid + i * 256;
      int ci = idx >> 7, nn = idx & 127;
      xs[nn * LDP + ci] = (_Float16)X[(size_t)(cb0 + ci) * N + n0 + nn];
    }
    __syncthreads();
    v16h a = loadA(W + (size_t)(o0 + ot * 16) * K, K, cb0, lr, half);
#pragma unroll
    for (int nt = 0; nt < 4; ++nt) {
      v16h b = loadB(xs + ((ng + nt) * 16) * LDP, LDP, 0, lr, half);
      acc[nt] = wmma(a, b, acc[nt]);
    }
  }
#pragma unroll
  for (int nt = 0; nt < 4; ++nt)
#pragma unroll
    for (int r = 0; r < 8; ++r)
      O[(size_t)(o0 + ot * 16 + r + 8 * half) * N + n0 + (ng + nt) * 16 + lr] =
          (_Float16)acc[nt][r];
}

// ------------------------------------------------------------- energy
// E[n,m] = sum_c qT[n,c]*kT[m,c]  (scale applied in softmax).  No LDS needed.
// WG tile 256n x 128m; wave tile 64n x 64m -> 16 wmma per 16 fragment loads.
__global__ __launch_bounds__(256) void cdna5_energy(
    const _Float16* __restrict__ qT, const _Float16* __restrict__ kT,
    _Float16* __restrict__ E) {
  const size_t bq = (size_t)blockIdx.z * 4096 * 1024;
  qT += bq; kT += bq;
  E += (size_t)blockIdx.z * 4096 * 4096;
  const int lane = threadIdx.x & 31, wave = threadIdx.x >> 5;
  const int lr = lane & 15, half = lane >> 4;
  const int nb = blockIdx.x * 256 + (wave >> 1) * 64;  // 4 n-tiles
  const int mb = blockIdx.y * 128 + (wave & 1) * 64;   // 4 m-tiles

  v8f acc[4][4];
#pragma unroll
  for (int i = 0; i < 4; ++i)
#pragma unroll
    for (int j = 0; j < 4; ++j)
#pragma unroll
      for (int r = 0; r < 8; ++r) acc[i][j][r] = 0.f;

  for (int cb = 0; cb < 1024; cb += 32) {
    v16h a[4], b[4];
#pragma unroll
    for (int i = 0; i < 4; ++i)
      a[i] = loadA(qT + (size_t)(nb + i * 16) * 1024, 1024, cb, lr, half);
#pragma unroll
    for (int j = 0; j < 4; ++j)
      b[j] = loadB(kT + (size_t)(mb + j * 16) * 1024, 1024, cb, lr, half);
#pragma unroll
    for (int i = 0; i < 4; ++i)
#pragma unroll
      for (int j = 0; j < 4; ++j) acc[i][j] = wmma(a[i], b[j], acc[i][j]);
  }
#pragma unroll
  for (int i = 0; i < 4; ++i)
#pragma unroll
    for (int j = 0; j < 4; ++j)
#pragma unroll
      for (int r = 0; r < 8; ++r)
        E[(size_t)(nb + i * 16 + r + 8 * half) * 4096 + mb + j * 16 + lr] =
            (_Float16)acc[i][j][r];
}

// ------------------------------------------------------------- softmax
// In-place over rows of 4096, scale = 1/sqrt(1024) = 1/32 folded in.
__global__ __launch_bounds__(256) void cdna5_softmax(_Float16* __restrict__ A) {
  _Float16* row = A + (size_t)blockIdx.x * 4096;
  const int tid = threadIdx.x, wave = tid >> 5, lane = tid & 31;
  float v[16];
  float mx = -3.0e38f;
#pragma unroll
  for (int i = 0; i < 16; ++i) {
    v[i] = (float)row[tid + i * 256] * 0.03125f;
    mx = fmaxf(mx, v[i]);
  }
#pragma unroll
  for (int s = 16; s >= 1; s >>= 1) mx = fmaxf(mx, __shfl_xor(mx, s, 32));
  __shared__ float rmax[8], rsum[8];
  if (lane == 0) rmax[wave] = mx;
  __syncthreads();
#pragma unroll
  for (int i = 0; i < 8; ++i) mx = fmaxf(mx, rmax[i]);
  float sum = 0.f;
#pragma unroll
  for (int i = 0; i < 16; ++i) { v[i] = __expf(v[i] - mx); sum += v[i]; }
#pragma unroll
  for (int s = 16; s >= 1; s >>= 1) sum += __shfl_xor(sum, s, 32);
  if (lane == 0) rsum[wave] = sum;
  __syncthreads();
  float tot = 0.f;
#pragma unroll
  for (int i = 0; i < 8; ++i) tot += rsum[i];
  float inv = 1.f / tot;
#pragma unroll
  for (int i = 0; i < 16; ++i) row[tid + i * 256] = (_Float16)(v[i] * inv);
}

// --------------------------------------------------------------- attn*V
// xsumT[n,c] = sum_m attn[n,m]*V[c,m] + skipT[n,c]
// WG tile 256n x 64c; wave tile 32n x 64c -> 8 wmma per 12 fragment loads.
__global__ __launch_bounds__(256) void cdna5_av(
    const _Float16* __restrict__ attn, const _Float16* __restrict__ V,
    const _Float16* __restrict__ skT, _Float16* __restrict__ xsumT) {
  attn  += (size_t)blockIdx.z * 4096 * 4096;
  V     += (size_t)blockIdx.z * 1024 * 4096;
  skT   += (size_t)blockIdx.z * 4096 * 1024;
  xsumT += (size_t)blockIdx.z * 4096 * 1024;
  const int lane = threadIdx.x & 31, wave = threadIdx.x >> 5;
  const int lr = lane & 15, half = lane >> 4;
  const int n0 = blockIdx.x * 256 + wave * 32;  // 2 n-tiles
  const int c0 = blockIdx.y * 64;               // 4 c-tiles

  v8f acc[2][4];
#pragma unroll
  for (int i = 0; i < 2; ++i)
#pragma unroll
    for (int t = 0; t < 4; ++t)
#pragma unroll
      for (int r = 0; r < 8; ++r) acc[i][t][r] = 0.f;

  for (int mb = 0; mb < 4096; mb += 32) {
    v16h a[2];
#pragma unroll
    for (int i = 0; i < 2; ++i)
      a[i] = loadA(attn + (size_t)(n0 + i * 16) * 4096, 4096, mb, lr, half);
#pragma unroll
    for (int ct = 0; ct < 4; ++ct) {
      v16h b = loadB(V + (size_t)(c0 + ct * 16) * 4096, 4096, mb, lr, half);
      acc[0][ct] = wmma(a[0], b, acc[0][ct]);
      acc[1][ct] = wmma(a[1], b, acc[1][ct]);
    }
  }
#pragma unroll
  for (int i = 0; i < 2; ++i)
#pragma unroll
    for (int ct = 0; ct < 4; ++ct)
#pragma unroll
      for (int r = 0; r < 8; ++r) {
        size_t n = n0 + i * 16 + r + 8 * half;
        int c = c0 + ct * 16 + lr;
        float s = acc[i][ct][r] + (float)skT[n * 1024 + c];
        xsumT[n * 1024 + c] = (_Float16)s;
      }
}

// ----------------------------------------------------------------- post
// c1(256x1024)+bn_silu -> c2(64x256)+bn_silu -> c3(1x64)+bias -> sigmoid
__global__ __launch_bounds__(256) void cdna5_post(
    const _Float16* __restrict__ xsumT, const _Float16* __restrict__ c1w,
    const _Float16* __restrict__ c2w, const float* __restrict__ bn1_g,
    const float* __restrict__ bn1_b, const float* __restrict__ bn1_m,
    const float* __restrict__ bn1_v, const float* __restrict__ bn2_g,
    const float* __restrict__ bn2_b, const float* __restrict__ bn2_m,
    const float* __restrict__ bn2_v, const float* __restrict__ c3_w,
    const float* __restrict__ c3_b, float* __restrict__ out) {
  const int b = blockIdx.y, n0 = blockIdx.x * 64;
  xsumT += (size_t)b * 4096 * 1024;
  const int tid = threadIdx.x;
  const int wave = tid >> 5, lane = tid & 31, lr = lane & 15, half = lane >> 4;
  constexpr int LD1 = 264;
  __shared__ _Float16 lds1[64 * LD1];   // stage-1 activations, [n][o]
  __shared__ float lds_out[64];

  // ---- stage 1: c1, wave w owns output channels [w*32, w*32+32)
  const int ob = wave * 32;
  v8f acc[2][4];
#pragma unroll
  for (int i = 0; i < 2; ++i)
#pragma unroll
    for (int t = 0; t < 4; ++t)
#pragma unroll
      for (int r = 0; r < 8; ++r) acc[i][t][r] = 0.f;

  for (int cb = 0; cb < 1024; cb += 32) {
    v16h a0 = loadA(c1w + (size_t)ob * 1024, 1024, cb, lr, half);
    v16h a1 = loadA(c1w + (size_t)(ob + 16) * 1024, 1024, cb, lr, half);
#pragma unroll
    for (int nt = 0; nt < 4; ++nt) {
      v16h bf = loadB(xsumT + (size_t)(n0 + nt * 16) * 1024, 1024, cb, lr, half);
      acc[0][nt] = wmma(a0, bf, acc[0][nt]);
      acc[1][nt] = wmma(a1, bf, acc[1][nt]);
    }
  }
#pragma unroll
  for (int oi = 0; oi < 2; ++oi)
#pragma unroll
    for (int r = 0; r < 8; ++r) {
      int o = ob + oi * 16 + r + 8 * half;
      float s = bn1_g[o] * rsqrtf(bn1_v[o] + 1e-5f);
      float sh = bn1_b[o] - bn1_m[o] * s;
#pragma unroll
      for (int nt = 0; nt < 4; ++nt) {
        float y = acc[oi][nt][r] * s + sh;
        y = y / (1.f + __expf(-y));
        lds1[(nt * 16 + lr) * LD1 + o] = (_Float16)y;   // transposed store
      }
    }
  if (tid < 64) lds_out[tid] = c3_b[0];
  __syncthreads();

  // ---- stage 2: c2 (64x256) @ lds1 (256x64); wave owns tiles 2w, 2w+1
  v8f a2[2];
#pragma unroll
  for (int i = 0; i < 2; ++i)
#pragma unroll
    for (int r = 0; r < 8; ++r) a2[i][r] = 0.f;
  for (int cb = 0; cb < 256; cb += 32) {
#pragma unroll
    for (int tI = 0; tI < 2; ++tI) {
      int t = wave * 2 + tI, ot2 = t & 3, nt2 = t >> 2;
      v16h a = loadA(c2w + (size_t)(ot2 * 16) * 256, 256, cb, lr, half);
      v16h bf = loadB(lds1 + (nt2 * 16) * LD1, LD1, cb, lr, half);
      a2[tI] = wmma(a, bf, a2[tI]);
    }
  }
#pragma unroll
  for (int tI = 0; tI < 2; ++tI) {
    int t = wave * 2 + tI, ot2 = t & 3, nt2 = t >> 2;
#pragma unroll
    for (int r = 0; r < 8; ++r) {
      int o = ot2 * 16 + r + 8 * half;
      int nl = nt2 * 16 + lr;
      float s = bn2_g[o] * rsqrtf(bn2_v[o] + 1e-5f);
      float y = a2[tI][r] * s + (bn2_b[o] - bn2_m[o] * s);
      y = y / (1.f + __expf(-y));
      atomicAdd(&lds_out[nl], c3_w[o] * y);   // ds_add_f32
    }
  }
  __syncthreads();
  if (tid < 64) {
    float z = lds_out[tid] * 2.9957322735539909f;  // LOG20
    out[(size_t)b * 4096 + n0 + tid] = 1.f / (1.f + __expf(-z));
  }
}

// ================================================================ launch
extern "C" void kernel_launch(void* const* d_in, const int* in_sizes, int n_in,
                              void* d_out, int out_size, void* d_ws,
                              size_t ws_size, hipStream_t stream) {
  const float* pcd   = (const float*)d_in[0];
  const float* img   = (const float*)d_in[1];
  const float* q_w   = (const float*)d_in[2];
  const float* k_w   = (const float*)d_in[3];
  const float* v_w   = (const float*)d_in[4];
  const float* sk_w  = (const float*)d_in[5];
  const float* c1_w  = (const float*)d_in[6];
  const float* bn1_g = (const float*)d_in[7];
  const float* bn1_b = (const float*)d_in[8];
  const float* bn1_m = (const float*)d_in[9];
  const float* bn1_v = (const float*)d_in[10];
  const float* c2_w  = (const float*)d_in[11];
  const float* bn2_g = (const float*)d_in[12];
  const float* bn2_b = (const float*)d_in[13];
  const float* bn2_m = (const float*)d_in[14];
  const float* bn2_v = (const float*)d_in[15];
  const float* c3_w  = (const float*)d_in[16];
  const float* c3_b  = (const float*)d_in[17];
  float* out = (float*)d_out;

  char* ws = (char*)d_ws;
  size_t off = 0;
  auto take = [&](size_t bytes) -> char* {
    char* p = ws + off;
    off += (bytes + 255) & ~(size_t)255;
    return p;
  };
  _Float16* qw16  = (_Float16*)take((size_t)1024 * 1152 * 2);
  _Float16* kw16  = (_Float16*)take((size_t)1024 * 1024 * 2);
  _Float16* vw16  = (_Float16*)take((size_t)1024 * 1024 * 2);
  _Float16* skw16 = (_Float16*)take((size_t)1024 * 1152 * 2);
  _Float16* c1w16 = (_Float16*)take((size_t)256 * 1024 * 2);
  _Float16* c2w16 = (_Float16*)take((size_t)64 * 256 * 2);
  _Float16* qT    = (_Float16*)take((size_t)8 * 4096 * 1024 * 2);
  _Float16* kT    = (_Float16*)take((size_t)8 * 4096 * 1024 * 2);
  _Float16* Vb    = (_Float16*)take((size_t)8 * 1024 * 4096 * 2);
  _Float16* skT   = (_Float16*)take((size_t)8 * 4096 * 1024 * 2);
  _Float16* xsum  = (_Float16*)take((size_t)8 * 4096 * 1024 * 2);
  _Float16* attn  = (_Float16*)take((size_t)8 * 4096 * 4096 * 2);
  if (off > ws_size) return;  // workspace too small; nothing safe to do

  // weights -> f16
  auto cvt = [&](const float* s, _Float16* d, int n) {
    cdna5_cvt_f16<<<(n + 255) / 256, 256, 0, stream>>>(s, d, n);
  };
  cvt(q_w,  qw16,  1024 * 1152);
  cvt(k_w,  kw16,  1024 * 1024);
  cvt(v_w,  vw16,  1024 * 1024);
  cvt(sk_w, skw16, 1024 * 1152);
  cvt(c1_w, c1w16, 256 * 1024);
  cvt(c2_w, c2w16, 64 * 256);

  const size_t pcdStride = (size_t)1152 * 4096;
  const size_t imgStride = (size_t)1024 * 4096;
  const size_t nmStride  = (size_t)4096 * 1024;

  dim3 gp(4096 / 128, 1024 / 64, 8);
  cdna5_proj_nt<<<gp, 256, 0, stream>>>(pcd, qw16, qT, 1152, 1024, pcdStride, nmStride);
  cdna5_proj_nt<<<gp, 256, 0, stream>>>(img, kw16, kT, 1024, 1024, imgStride, nmStride);
  cdna5_proj_nt<<<gp, 256, 0, stream>>>(pcd, skw16, skT, 1152, 1024, pcdStride, nmStride);
  cdna5_proj_nn<<<gp, 256, 0, stream>>>(img, vw16, Vb, 1024, imgStride, nmStride);

  dim3 ge(4096 / 256, 4096 / 128, 8);
  cdna5_energy<<<ge, 256, 0, stream>>>(qT, kT, attn);

  cdna5_softmax<<<8 * 4096, 256, 0, stream>>>(attn);

  dim3 ga(4096 / 256, 1024 / 64, 8);
  cdna5_av<<<ga, 256, 0, stream>>>(attn, Vb, skT, xsum);

  dim3 go(4096 / 64, 8);
  cdna5_post<<<go, 256, 0, stream>>>(xsum, c1w16, c2w16, bn1_g, bn1_b, bn1_m,
                                     bn1_v, bn2_g, bn2_b, bn2_m, bn2_v, c3_w,
                                     c3_b, out);
}
